// GCN_more_layer_SGD_58248346468472
// MI455X (gfx1250) — compile-verified, matrix operand order checked by reading, not run on hardware
//
#include <hip/hip_runtime.h>

// ---------------------------------------------------------------------------
// GCN 3-layer forward for MI455X (gfx1250, wave32).
//  - GEMMs via v_wmma_f32_16x16x32_f16; one wave owns a 16x128 output tile.
//    Packed B (weights) staged in LDS per block (32KB of the 320KB WGP LDS),
//    B loads batched per k-step so 8 WMMAs issue back-to-back.
//  - edge aggregation via hardware f32 atomics (working set ~90MB, fully
//    resident in the 192MB L2).
// ---------------------------------------------------------------------------

typedef __attribute__((ext_vector_type(16))) _Float16 v16h;
typedef __attribute__((ext_vector_type(8)))  float    v8f;

#define C_DIM  128
#define KSTEPS 4    // 128 / 32
#define NTILES 8    // 128 / 16
#define WTILE_VECS (NTILES * KSTEPS * 32)   // v16h elements in packed W

// ---------------- degree / normalization ----------------
__global__ void k_deg_init(float* __restrict__ deg, int n) {
  int i = blockIdx.x * blockDim.x + threadIdx.x;
  if (i < n) deg[i] = 1.0f;                         // self-loop contribution
}

__global__ void k_deg_edges(const long long* __restrict__ ei,
                            float* __restrict__ deg, int e_cnt) {
  int e = blockIdx.x * blockDim.x + threadIdx.x;
  if (e < e_cnt) {
    int c = (int)ei[(size_t)e_cnt + e];             // col = target
    unsafeAtomicAdd(&deg[c], 1.0f);                 // integer-valued: exact
  }
}

__global__ void k_rsqrt_inplace(float* __restrict__ d, int n) {
  int i = blockIdx.x * blockDim.x + threadIdx.x;
  if (i < n) { float v = d[i]; d[i] = (v > 0.f) ? rsqrtf(v) : 0.f; }
}

// ---------------- WMMA operand packing ----------------
// A layout (16-bit A 16x32, ISA 7.12.2): lane L holds row M = L%16,
// h = L/16 selects K-halves: regs j=0..7 -> K = k0+8h+j, j=8..15 -> K = k0+16+8h+(j-8).
__global__ void k_pack_act(const float* __restrict__ act,
                           _Float16* __restrict__ packA, int mtiles) {
  int idx = blockIdx.x * blockDim.x + threadIdx.x;  // (tile, s, lane)
  if (idx >= mtiles * KSTEPS * 32) return;
  int lane = idx & 31;
  int s    = (idx >> 5) & (KSTEPS - 1);
  int tile = idx >> 7;
  int m  = tile * 16 + (lane & 15);
  int h  = lane >> 4;
  int k0 = s * 32;
  const float* src = act + (size_t)m * C_DIM;
  _Float16* dst = packA + (size_t)idx * 16;
#pragma unroll
  for (int j = 0; j < 8; ++j) dst[j]     = (_Float16)src[k0 + 8 * h + j];
#pragma unroll
  for (int j = 0; j < 8; ++j) dst[8 + j] = (_Float16)src[k0 + 16 + 8 * h + j];
}

// B layout mirrored: lane L holds column N = L%16, same K pattern. W row-major [K][N].
__global__ void k_pack_w(const float* __restrict__ W, _Float16* __restrict__ packW) {
  int idx = blockIdx.x * blockDim.x + threadIdx.x;  // (ntile, s, lane)
  if (idx >= WTILE_VECS) return;
  int lane = idx & 31;
  int s    = (idx >> 5) & (KSTEPS - 1);
  int nt   = idx >> 7;
  int n  = nt * 16 + (lane & 15);
  int h  = lane >> 4;
  int k0 = s * 32;
  _Float16* dst = packW + (size_t)idx * 16;
#pragma unroll
  for (int j = 0; j < 8; ++j) dst[j]     = (_Float16)W[(size_t)(k0 + 8 * h + j) * C_DIM + n];
#pragma unroll
  for (int j = 0; j < 8; ++j) dst[8 + j] = (_Float16)W[(size_t)(k0 + 16 + 8 * h + j) * C_DIM + n];
}

// ---------------- WMMA GEMM: H[mtile*16 .. +16, 0..128) = A @ W ----------------
__global__ void __launch_bounds__(256)
k_gemm_wmma(const v16h* __restrict__ packA,   // [mtiles][KSTEPS][32]
            const v16h* __restrict__ packW,   // [NTILES][KSTEPS][32]
            float* __restrict__ H, int mtiles) {
  // stage packed weights (32KB) in LDS: shared by all 8 waves of the block
  __shared__ v16h sW[WTILE_VECS];
  {
    const uint4* gsrc = (const uint4*)packW;
    uint4*       ldst = (uint4*)sW;
    for (int i = threadIdx.x; i < WTILE_VECS * 2; i += blockDim.x)
      ldst[i] = gsrc[i];
  }
  __syncthreads();

  int wave = (int)((blockIdx.x * blockDim.x + threadIdx.x) >> 5);
  int lane = threadIdx.x & 31;
  if (wave >= mtiles) return;

  v8f acc[NTILES] = {};
#pragma unroll
  for (int s = 0; s < KSTEPS; ++s) {
    v16h a = packA[((size_t)wave * KSTEPS + s) * 32 + lane];
    v16h b[NTILES];
#pragma unroll
    for (int nt = 0; nt < NTILES; ++nt)
      b[nt] = sW[(nt * KSTEPS + s) * 32 + lane];
#pragma unroll
    for (int nt = 0; nt < NTILES; ++nt)
      acc[nt] = __builtin_amdgcn_wmma_f32_16x16x32_f16(
          /*neg_a=*/false, a, /*neg_b=*/false, b[nt],
          /*c_mod=*/(short)0, acc[nt], /*reuse_a=*/false, /*reuse_b=*/false);
  }

  // C/D layout: VGPR r -> M = r + 8*(lane/16), N = 16*nt + lane%16
  int h   = lane >> 4;
  int nlo = lane & 15;
  size_t mbase = (size_t)wave * 16;
#pragma unroll
  for (int nt = 0; nt < NTILES; ++nt) {
#pragma unroll
    for (int r = 0; r < 8; ++r) {
      H[(mbase + r + 8 * h) * C_DIM + nt * 16 + nlo] = acc[nt][r];
    }
  }
}

// ---------------- aggregation ----------------
// agg = Hlin * dinv^2  (self-loop term; also fully initializes the buffer)
__global__ void k_init_agg(const float* __restrict__ Hlin,
                           const float* __restrict__ dinv,
                           float* __restrict__ agg, int total) {
  int i = blockIdx.x * blockDim.x + threadIdx.x;
  if (i < total) {
    float di = dinv[i >> 7];
    agg[i] = Hlin[i] * di * di;
  }
}

// one wave per edge: lane f4-gathers 4 features, 4 hardware f32 atomics
__global__ void __launch_bounds__(256)
k_edge_scatter(const long long* __restrict__ ei,
               const float* __restrict__ Hlin,
               const float* __restrict__ dinv,
               float* __restrict__ agg, int e_cnt) {
  int gid  = blockIdx.x * blockDim.x + threadIdx.x;
  int e    = gid >> 5;
  int lane = gid & 31;
  if (e >= e_cnt) return;
  int r = (int)ei[e];
  int c = (int)ei[(size_t)e_cnt + e];
  float w = dinv[r] * dinv[c];
  const float4* src = (const float4*)(Hlin + (size_t)r * C_DIM);
  float4 v = src[lane];
  float* dst = agg + (size_t)c * C_DIM + 4 * lane;
  unsafeAtomicAdd(dst + 0, v.x * w);
  unsafeAtomicAdd(dst + 1, v.y * w);
  unsafeAtomicAdd(dst + 2, v.z * w);
  unsafeAtomicAdd(dst + 3, v.w * w);
}

__global__ void k_bias_relu(const float* __restrict__ agg,
                            const float* __restrict__ b,
                            float* __restrict__ act, int total) {
  int i = blockIdx.x * blockDim.x + threadIdx.x;
  if (i < total) {
    float v = agg[i] + b[i & (C_DIM - 1)];
    act[i] = v > 0.f ? v : 0.f;
  }
}

// ---------------- layer 3 (128 -> 1) ----------------
__global__ void k_gemv_out(const float* __restrict__ act,
                           const float* __restrict__ W3,
                           float* __restrict__ h3, int n) {
  int i = blockIdx.x * blockDim.x + threadIdx.x;
  if (i >= n) return;
  const float4* a = (const float4*)(act + (size_t)i * C_DIM);
  const float4* w = (const float4*)W3;
  float s = 0.f;
#pragma unroll
  for (int j = 0; j < C_DIM / 4; ++j) {
    float4 av = a[j], wv = w[j];
    s += av.x * wv.x + av.y * wv.y + av.z * wv.z + av.w * wv.w;
  }
  h3[i] = s;
}

__global__ void k_init_out3(const float* __restrict__ h3,
                            const float* __restrict__ dinv,
                            float* __restrict__ out3, int n) {
  int i = blockIdx.x * blockDim.x + threadIdx.x;
  if (i < n) { float di = dinv[i]; out3[i] = h3[i] * di * di; }
}

__global__ void k_edge_scatter_scalar(const long long* __restrict__ ei,
                                      const float* __restrict__ h3,
                                      const float* __restrict__ dinv,
                                      float* __restrict__ out3, int e_cnt) {
  int e = blockIdx.x * blockDim.x + threadIdx.x;
  if (e >= e_cnt) return;
  int r = (int)ei[e];
  int c = (int)ei[(size_t)e_cnt + e];
  unsafeAtomicAdd(&out3[c], h3[r] * dinv[r] * dinv[c]);
}

__global__ void k_sigmoid(const float* __restrict__ out3,
                          const float* __restrict__ b3,
                          float* __restrict__ out, int n) {
  int i = blockIdx.x * blockDim.x + threadIdx.x;
  if (i < n) {
    float z = out3[i] + b3[0];
    out[i] = 1.f / (1.f + __expf(-z));
  }
}

// ---------------- host orchestration ----------------
extern "C" void kernel_launch(void* const* d_in, const int* in_sizes, int n_in,
                              void* d_out, int out_size, void* d_ws, size_t ws_size,
                              hipStream_t stream) {
  const float*     x  = (const float*)d_in[0];
  const long long* ei = (const long long*)d_in[1];
  const float*     W1 = (const float*)d_in[2];
  const float*     b1 = (const float*)d_in[3];
  const float*     W2 = (const float*)d_in[4];
  const float*     b2 = (const float*)d_in[5];
  const float*     W3 = (const float*)d_in[6];
  const float*     b3 = (const float*)d_in[7];
  float* out = (float*)d_out;

  const int N = in_sizes[0] / C_DIM;      // 50000
  const int E = in_sizes[1] / 2;          // 800000
  const int mtiles = (N + 15) / 16;       // 3125 (N divisible by 16)
  const int total  = N * C_DIM;

  // workspace carve-out (256B aligned)
  char* ws = (char*)d_ws;
  size_t off = 0;
  auto carve = [&](size_t bytes) -> char* {
    size_t o = (off + 255) & ~(size_t)255;
    off = o + bytes;
    return ws + o;
  };
  float*    dinv  = (float*)   carve((size_t)N * 4);
  float*    Hlin  = (float*)   carve((size_t)total * 4);
  float*    agg   = (float*)   carve((size_t)total * 4);
  float*    act   = (float*)   carve((size_t)total * 4);
  _Float16* packA = (_Float16*)carve((size_t)mtiles * KSTEPS * 32 * 16 * 2);
  _Float16* packW = (_Float16*)carve((size_t)WTILE_VECS * 16 * 2);
  float*    h3    = (float*)   carve((size_t)N * 4);
  float*    out3  = (float*)   carve((size_t)N * 4);
  (void)ws_size; (void)n_in; (void)out_size;

  const int B = 256;
  auto blk = [](long long n, int b) { return (unsigned)((n + b - 1) / b); };

  // normalization: deg -> dinv (in place)
  k_deg_init<<<blk(N, B), B, 0, stream>>>(dinv, N);
  k_deg_edges<<<blk(E, B), B, 0, stream>>>(ei, dinv, E);
  k_rsqrt_inplace<<<blk(N, B), B, 0, stream>>>(dinv, N);

  const int packA_th = mtiles * KSTEPS * 32;
  const int packW_th = WTILE_VECS;
  const long long edge_th = (long long)E * 32;
  const unsigned gemm_blocks = blk((long long)mtiles * 32, B);

  // ---- layer 1 ----
  k_pack_act<<<blk(packA_th, B), B, 0, stream>>>(x, packA, mtiles);
  k_pack_w  <<<blk(packW_th, B), B, 0, stream>>>(W1, packW);
  k_gemm_wmma<<<gemm_blocks, B, 0, stream>>>((const v16h*)packA, (const v16h*)packW, Hlin, mtiles);
  k_init_agg<<<blk(total, B), B, 0, stream>>>(Hlin, dinv, agg, total);
  k_edge_scatter<<<blk(edge_th, B), B, 0, stream>>>(ei, Hlin, dinv, agg, E);
  k_bias_relu<<<blk(total, B), B, 0, stream>>>(agg, b1, act, total);

  // ---- layer 2 ----
  k_pack_act<<<blk(packA_th, B), B, 0, stream>>>(act, packA, mtiles);
  k_pack_w  <<<blk(packW_th, B), B, 0, stream>>>(W2, packW);
  k_gemm_wmma<<<gemm_blocks, B, 0, stream>>>((const v16h*)packA, (const v16h*)packW, Hlin, mtiles);
  k_init_agg<<<blk(total, B), B, 0, stream>>>(Hlin, dinv, agg, total);
  k_edge_scatter<<<blk(edge_th, B), B, 0, stream>>>(ei, Hlin, dinv, agg, E);
  k_bias_relu<<<blk(total, B), B, 0, stream>>>(agg, b2, act, total);

  // ---- layer 3 (128 -> 1) + sigmoid ----
  k_gemv_out<<<blk(N, B), B, 0, stream>>>(act, W3, h3, N);
  k_init_out3<<<blk(N, B), B, 0, stream>>>(h3, dinv, out3, N);
  k_edge_scatter_scalar<<<blk(E, B), B, 0, stream>>>(ei, h3, dinv, out3, E);
  k_sigmoid<<<blk(N, B), B, 0, stream>>>(out3, b3, out, N);
}
